// GCRNCell_21586505629814
// MI455X (gfx1250) — matrix operand order repaired and checked by reading, not compiled, not run on hardware
//
#include <hip/hip_runtime.h>
#include <math.h>

// ---------------- problem constants ----------------
constexpr int B_   = 8;
constexpr int N_   = 2048;
constexpr int IN_  = 2;
constexpr int OUT_ = 16;
constexpr int C_   = 18;      // IN + OUT
constexpr int H_   = 4;
constexpr int DH_  = 8;
constexpr int BC_  = B_ * C_; // 144 GEMM columns
constexpr int NG_  = 6;       // 2 predefined + 4 adaptive heads

typedef __attribute__((ext_vector_type(16))) __bf16 v16bf;
typedef __attribute__((ext_vector_type(8)))  __bf16 v8bf;
typedef __attribute__((ext_vector_type(8)))  float  v8f;

union V16U { v16bf v; v8bf h[2]; };

// ---------------- kernel 1: f32 -> bf16 convert (predefined A) ----------------
__global__ void k_cvt_bf16(const float* __restrict__ src, __bf16* __restrict__ dst, size_t n) {
    size_t i = (size_t)blockIdx.x * blockDim.x + threadIdx.x;
    size_t stride = (size_t)gridDim.x * blockDim.x;
    for (; i < n; i += stride) dst[i] = (__bf16)src[i];
}

// ---------------- kernel 2: adaptive adjacency: softmax(relu(E1 E2^T)/sqrt(DH)) ----------------
// one block per (head, row); writes bf16 row of Aad[h][w][:]
__global__ __launch_bounds__(256)
void k_adaptive(const float* __restrict__ E1, const float* __restrict__ E2,
                __bf16* __restrict__ Aad) {
    const int h = blockIdx.x >> 11;       // / N_
    const int w = blockIdx.x & (N_ - 1);
    __shared__ float row[N_];
    __shared__ float red[256];

    float e1[DH_];
#pragma unroll
    for (int d = 0; d < DH_; ++d) e1[d] = E1[((size_t)w * H_ + h) * DH_ + d];

    const float scale = 0.3535533905932738f; // 1/sqrt(8)
    float lmax = 0.0f;                       // relu >= 0 so 0 is a valid floor
    for (int v = threadIdx.x; v < N_; v += 256) {
        float s = 0.f;
#pragma unroll
        for (int d = 0; d < DH_; ++d) s += e1[d] * E2[((size_t)v * H_ + h) * DH_ + d];
        s = fmaxf(s, 0.f) * scale;
        row[v] = s;
        lmax = fmaxf(lmax, s);
    }
    red[threadIdx.x] = lmax; __syncthreads();
    for (int s = 128; s > 0; s >>= 1) {
        if (threadIdx.x < s) red[threadIdx.x] = fmaxf(red[threadIdx.x], red[threadIdx.x + s]);
        __syncthreads();
    }
    const float m = red[0]; __syncthreads();

    float lsum = 0.f;
    for (int v = threadIdx.x; v < N_; v += 256) {
        float e = __expf(row[v] - m);
        row[v] = e;
        lsum += e;
    }
    red[threadIdx.x] = lsum; __syncthreads();
    for (int s = 128; s > 0; s >>= 1) {
        if (threadIdx.x < s) red[threadIdx.x] += red[threadIdx.x + s];
        __syncthreads();
    }
    const float inv = 1.f / red[0];

    __bf16* dst = Aad + ((size_t)h * N_ + w) * N_;
    for (int v = threadIdx.x; v < N_; v += 256) dst[v] = (__bf16)(row[v] * inv);
}

// ---------------- kernel 3: build ias (f32) and transposed bf16 GEMM operand ----------------
__global__ void k_prep(const float* __restrict__ x, const float* __restrict__ hid,
                       float* __restrict__ iasf, __bf16* __restrict__ Ht) {
    int idx = blockIdx.x * blockDim.x + threadIdx.x;   // over B*N
    if (idx >= B_ * N_) return;
    const int b = idx / N_, n = idx % N_;
    float v[C_];
    v[0] = x[(size_t)idx * IN_ + 0];
    v[1] = x[(size_t)idx * IN_ + 1];
#pragma unroll
    for (int c = 0; c < OUT_; ++c) v[2 + c] = hid[(size_t)idx * OUT_ + c];
#pragma unroll
    for (int c = 0; c < C_; ++c) {
        iasf[(size_t)idx * C_ + c] = v[c];
        Ht[((size_t)(b * C_ + c)) * N_ + n] = (__bf16)v[c];
    }
}

// ---------------- kernel 4: WMMA GEMM  Y[g] = A[g] @ H  ----------------
// A: [NG][N][N] bf16 row-major.  B operand: transposed [col][k] bf16, ld = N.
//   bStrideG = 0  -> all g share one B (pass 1)
//   bStrideG = BC_*N_ -> per-g B (pass 2, B = previous pass output)
// Output: Yf transposed f32 [g][col][node]; optional Yb bf16 same layout (feeds next pass).
// Wave tiling: wave = 16 output rows x 48 output cols (3 WMMA col tiles, A fragment reused).
__global__ __launch_bounds__(128)
void k_gemm(const __bf16* __restrict__ Aall, const __bf16* __restrict__ Ball,
            size_t bStrideG, float* __restrict__ Yf, __bf16* __restrict__ Yb) {
    const int g    = blockIdx.z;
    const int wave = threadIdx.x >> 5;
    const int lane = threadIdx.x & 31;
    const int rowTile  = blockIdx.x * 4 + wave;          // 0..127
    const int rowBase  = rowTile * 16;
    const int colGroup = blockIdx.y;                     // 0..2 -> col tiles 3*colGroup+{0,1,2}
    const int lh = lane >> 4;                            // 0 | 1
    const int ln = lane & 15;

    const __bf16* Ag = Aall + (size_t)g * N_ * N_;
    const __bf16* Bg = Ball + (size_t)g * bStrideG;

    // A fragment addressing (K-contiguous per lane half, same order as B):
    //   lane<16: row=ln, K = 0..15 ; lane>=16: row=ln, K = 16..31
    const __bf16* aPtr = Ag + (size_t)(rowBase + ln) * N_ + lh * 16;
    // B fragment addressing (32x16 bf16, K-contiguous per lane half):
    //   lane<16: col=ln, K = 0..15 ; lane>=16: col=ln, K = 16..31
    const __bf16* bPtr0 = Bg + (size_t)((colGroup * 3 + 0) * 16 + ln) * N_ + lh * 16;
    const __bf16* bPtr1 = bPtr0 + (size_t)16 * N_;
    const __bf16* bPtr2 = bPtr1 + (size_t)16 * N_;

    v8f zero;
#pragma unroll
    for (int i = 0; i < 8; ++i) zero[i] = 0.f;
    v8f acc0 = zero, acc1 = zero, acc2 = zero;

    for (int k0 = 0; k0 < N_; k0 += 32) {
        __builtin_prefetch(aPtr + k0 + 512, 0, 0);  // global_prefetch_b8, ~16 iters ahead
        V16U a;
        a.h[0] = *(const v8bf*)(aPtr + k0);
        a.h[1] = *(const v8bf*)(aPtr + k0 + 8);

        V16U b0, b1, b2;
        b0.h[0] = *(const v8bf*)(bPtr0 + k0);
        b0.h[1] = *(const v8bf*)(bPtr0 + k0 + 8);
        b1.h[0] = *(const v8bf*)(bPtr1 + k0);
        b1.h[1] = *(const v8bf*)(bPtr1 + k0 + 8);
        b2.h[0] = *(const v8bf*)(bPtr2 + k0);
        b2.h[1] = *(const v8bf*)(bPtr2 + k0 + 8);

        acc0 = __builtin_amdgcn_wmma_f32_16x16x32_bf16(false, a.v, false, b0.v, (short)0, acc0, false, false);
        acc1 = __builtin_amdgcn_wmma_f32_16x16x32_bf16(false, a.v, false, b1.v, (short)0, acc1, false, false);
        acc2 = __builtin_amdgcn_wmma_f32_16x16x32_bf16(false, a.v, false, b2.v, (short)0, acc2, false, false);
    }

    // D fragment: vgpr r holds (M = r + (lane>=16 ? 8:0), Ncol = ln).
    // Store transposed: Y[col][node] -> 8 consecutive nodes per lane.
    v8f accs[3] = {acc0, acc1, acc2};
#pragma unroll
    for (int t = 0; t < 3; ++t) {
        const int col = (colGroup * 3 + t) * 16 + ln;
        const size_t base = ((size_t)g * BC_ + col) * N_ + rowBase + lh * 8;
        float* yp = Yf + base;
        float4 s0 = {accs[t][0], accs[t][1], accs[t][2], accs[t][3]};
        float4 s1 = {accs[t][4], accs[t][5], accs[t][6], accs[t][7]};
        *(float4*)(yp)     = s0;
        *(float4*)(yp + 4) = s1;
        if (Yb) {
            v8bf bv;
#pragma unroll
            for (int r = 0; r < 8; ++r) bv[r] = (__bf16)accs[t][r];
            *(v8bf*)(Yb + base) = bv;
        }
    }
}

// helpers for the small gate linears: features are pre-reduced (sum/mean),
// g = a*(2*u0 + s1p + s2p + 2*bias) + (1-a)*(u0 + s1a + s2a + bias)
__device__ __forceinline__ float gate_out(const float* Wlds, const float* bias, float av,
                                          const float* f0, const float* f1p, const float* f2p,
                                          const float* f1a, const float* f2a, int oc) {
    float u0 = 0.f, s1p = 0.f, s2p = 0.f, s1a = 0.f, s2a = 0.f;
#pragma unroll
    for (int c = 0; c < C_; ++c) {
        const float w0 = Wlds[c * C_ + oc];
        const float w1 = Wlds[(C_ + c) * C_ + oc];
        const float w2 = Wlds[(2 * C_ + c) * C_ + oc];
        u0  += f0[c]  * w0;
        s1p += f1p[c] * w1;
        s2p += f2p[c] * w2;
        s1a += f1a[c] * w1;
        s2a += f2a[c] * w2;
    }
    const float bv = bias[oc];
    return av * (2.f * u0 + s1p + s2p + 2.f * bv) + (1.f - av) * (u0 + s1a + s2a + bv);
}

// gathers the 4 reduced feature vectors for one (b,n)
__device__ __forceinline__ void gather_feats(const float* __restrict__ Y1f, const float* __restrict__ Y2f,
                                             int b, int n, float* f1p, float* f2p, float* f1a, float* f2a) {
    constexpr size_t GS = (size_t)BC_ * N_;
#pragma unroll
    for (int c = 0; c < C_; ++c) {
        const size_t colOff = (size_t)(b * C_ + c) * N_ + n;
        f1p[c] = Y1f[colOff] + Y1f[GS + colOff];
        f2p[c] = Y2f[colOff] + Y2f[GS + colOff];
        float s1 = 0.f, s2 = 0.f;
#pragma unroll
        for (int h = 0; h < H_; ++h) {
            s1 += Y1f[(size_t)(2 + h) * GS + colOff];
            s2 += Y2f[(size_t)(2 + h) * GS + colOff];
        }
        f1a[c] = 0.25f * s1;
        f2a[c] = 0.25f * s2;
    }
}

// ---------------- kernel 5: z & r gates; emit r*ias (f32 + transposed bf16) ----------------
__global__ __launch_bounds__(256)
void k_combine_zr(const float* __restrict__ iasf,
                  const float* __restrict__ Y1f, const float* __restrict__ Y2f,
                  const float* __restrict__ gzW, const float* __restrict__ gzb,
                  const float* __restrict__ grW, const float* __restrict__ grb,
                  const float* __restrict__ aP,
                  float* __restrict__ zout, float* __restrict__ riasf, __bf16* __restrict__ H2t) {
    __shared__ float Wz[3 * C_ * C_];
    __shared__ float Wr[3 * C_ * C_];
    for (int i = threadIdx.x; i < 3 * C_ * C_; i += 256) { Wz[i] = gzW[i]; Wr[i] = grW[i]; }
    __syncthreads();

    const int idx = blockIdx.x * blockDim.x + threadIdx.x;
    if (idx >= B_ * N_) return;
    const int b = idx / N_, n = idx % N_;
    const float av = aP[0];

    float ias[C_], f1p[C_], f2p[C_], f1a[C_], f2a[C_];
#pragma unroll
    for (int c = 0; c < C_; ++c) ias[c] = iasf[(size_t)idx * C_ + c];
    gather_feats(Y1f, Y2f, b, n, f1p, f2p, f1a, f2a);

#pragma unroll
    for (int oc = 0; oc < C_; ++oc) {
        const float gz = gate_out(Wz, gzb, av, ias, f1p, f2p, f1a, f2a, oc);
        zout[(size_t)idx * C_ + oc] = 1.f / (1.f + __expf(-gz));
        const float gr = gate_out(Wr, grb, av, ias, f1p, f2p, f1a, f2a, oc);
        const float r  = 1.f / (1.f + __expf(-gr));
        const float rv = r * ias[oc];
        riasf[(size_t)idx * C_ + oc] = rv;
        H2t[(size_t)(b * C_ + oc) * N_ + n] = (__bf16)rv;
    }
}

// ---------------- kernel 6: c gate, GRU blend, output MLP ----------------
__global__ __launch_bounds__(256)
void k_final(const float* __restrict__ iasf, const float* __restrict__ riasf,
             const float* __restrict__ zbuf,
             const float* __restrict__ Y1f, const float* __restrict__ Y2f,
             const float* __restrict__ gcW, const float* __restrict__ gcb,
             const float* __restrict__ mlpW, const float* __restrict__ mlpb,
             const float* __restrict__ aP, float* __restrict__ out) {
    __shared__ float Wc[3 * C_ * C_];
    __shared__ float Wm[C_ * OUT_];
    for (int i = threadIdx.x; i < 3 * C_ * C_; i += 256) Wc[i] = gcW[i];
    for (int i = threadIdx.x; i < C_ * OUT_; i += 256) Wm[i] = mlpW[i];
    __syncthreads();

    const int idx = blockIdx.x * blockDim.x + threadIdx.x;
    if (idx >= B_ * N_) return;
    const int b = idx / N_, n = idx % N_;
    const float av = aP[0];

    float rias[C_], f1p[C_], f2p[C_], f1a[C_], f2a[C_];
#pragma unroll
    for (int c = 0; c < C_; ++c) rias[c] = riasf[(size_t)idx * C_ + c];
    gather_feats(Y1f, Y2f, b, n, f1p, f2p, f1a, f2a);

    float hvec[C_];
#pragma unroll
    for (int oc = 0; oc < C_; ++oc) {
        const float gc = gate_out(Wc, gcb, av, rias, f1p, f2p, f1a, f2a, oc);
        const float cv = tanhf(gc);
        const float z  = zbuf[(size_t)idx * C_ + oc];
        hvec[oc] = (1.f - z) * iasf[(size_t)idx * C_ + oc] + z * cv;
    }
#pragma unroll
    for (int o = 0; o < OUT_; ++o) {
        float s = mlpb[o];
#pragma unroll
        for (int c = 0; c < C_; ++c) s += hvec[c] * Wm[c * OUT_ + o];
        out[(size_t)idx * OUT_ + o] = s;
    }
}

// ---------------- launch ----------------
extern "C" void kernel_launch(void* const* d_in, const int* in_sizes, int n_in,
                              void* d_out, int out_size, void* d_ws, size_t ws_size,
                              hipStream_t stream) {
    const float* x    = (const float*)d_in[0];
    const float* hid  = (const float*)d_in[1];
    const float* preA = (const float*)d_in[2];
    const float* E1   = (const float*)d_in[3];
    const float* E2   = (const float*)d_in[4];
    const float* gzW  = (const float*)d_in[5];
    const float* gzb  = (const float*)d_in[6];
    const float* grW  = (const float*)d_in[7];
    const float* grb  = (const float*)d_in[8];
    const float* gcW  = (const float*)d_in[9];
    const float* gcb  = (const float*)d_in[10];
    const float* mlpW = (const float*)d_in[11];
    const float* mlpb = (const float*)d_in[12];
    const float* aP   = (const float*)d_in[13];
    float* out = (float*)d_out;

    // workspace carve (all chunk sizes are 256B-aligned by construction)
    constexpr size_t SZ_ABF = (size_t)NG_ * N_ * N_;   // bf16 elems
    constexpr size_t SZ_IAS = (size_t)B_ * N_ * C_;    // f32 elems
    constexpr size_t SZ_H   = (size_t)BC_ * N_;        // bf16 elems
    constexpr size_t SZ_Y   = (size_t)NG_ * BC_ * N_;  // elems

    char* p = (char*)d_ws;
    __bf16* Abf  = (__bf16*)p; p += SZ_ABF * 2;
    float*  iasf = (float*)p;  p += SZ_IAS * 4;
    float*  zbuf = (float*)p;  p += SZ_IAS * 4;
    float*  riasf= (float*)p;  p += SZ_IAS * 4;
    __bf16* H1t  = (__bf16*)p; p += SZ_H * 2;
    __bf16* H2t  = (__bf16*)p; p += SZ_H * 2;
    float*  Y1f  = (float*)p;  p += SZ_Y * 4;
    float*  Y2f  = (float*)p;  p += SZ_Y * 4;
    __bf16* Y1b  = (__bf16*)p; p += SZ_Y * 2;   // total ~69.4 MB

    // 1) predefined adjacencies -> bf16 (g = 0,1)
    k_cvt_bf16<<<2048, 256, 0, stream>>>(preA, Abf, (size_t)2 * N_ * N_);
    // 2) adaptive softmax adjacencies -> bf16 (g = 2..5)
    k_adaptive<<<H_ * N_, 256, 0, stream>>>(E1, E2, Abf + (size_t)2 * N_ * N_);
    // 3) ias (f32) + transposed bf16 GEMM operand
    k_prep<<<(B_ * N_ + 255) / 256, 256, 0, stream>>>(x, hid, iasf, H1t);

    const dim3 gGrid(32, 3, NG_); // 128 row tiles / 4 waves, 3 col groups, 6 matrices
    // 4,5) propagation for ias (shared by z and r gates)
    k_gemm<<<gGrid, 128, 0, stream>>>(Abf, H1t, 0, Y1f, Y1b);
    k_gemm<<<gGrid, 128, 0, stream>>>(Abf, Y1b, (size_t)BC_ * N_, Y2f, nullptr);
    // 6) z, r gates; build r*ias
    k_combine_zr<<<(B_ * N_ + 255) / 256, 256, 0, stream>>>(iasf, Y1f, Y2f, gzW, gzb, grW, grb,
                                                            aP, zbuf, riasf, H2t);
    // 7,8) propagation for r*ias (c gate)
    k_gemm<<<gGrid, 128, 0, stream>>>(Abf, H2t, 0, Y1f, Y1b);
    k_gemm<<<gGrid, 128, 0, stream>>>(Abf, Y1b, (size_t)BC_ * N_, Y2f, nullptr);
    // 9) c gate, GRU blend, output MLP
    k_final<<<(B_ * N_ + 255) / 256, 256, 0, stream>>>(iasf, riasf, zbuf, Y1f, Y2f,
                                                       gcW, gcb, mlpW, mlpb, aP, out);
}
